// BERTSelfAttention_14637248544808
// MI455X (gfx1250) — matrix operand order repaired and checked by reading, not compiled, not run on hardware
//
#include <hip/hip_runtime.h>
#include <hip/hip_bf16.h>

// ---------------------------------------------------------------------------
// BERT self-attention for MI455X (gfx1250): bf16 WMMA, flash-attention style.
// Round 3: softmax row-sums computed by WMMA (P x ones) instead of lane
// shuffles; 1/sqrt(DH)*log2(e) folded into Q at pack time so softmax is
// pure add/max/exp2. 16-bit WMMA co-executes with VALU (ISA: tracked as
// TRANS), so cutting softmax VALU directly shortens the critical path.
// ---------------------------------------------------------------------------

typedef __attribute__((ext_vector_type(16))) __bf16 v16bf;
typedef __attribute__((ext_vector_type(8)))  __bf16 v8bf;
typedef __attribute__((ext_vector_type(8)))  float  v8f;

#define H_   12
#define DH_  64
#define DM_  768
#define B_   2
#define S_   2048
#define NT_  (B_ * S_)          // 4096 tokens

#define LOG2E_   1.44269504088896f
#define SCALE_Q_ (0.125f * LOG2E_)   // 1/sqrt(DH) * log2(e), folded into Q

// ---- WMMA helpers ---------------------------------------------------------

__device__ __forceinline__ v8f wmma_bf16(v16bf a, v16bf b, v8f c) {
  // D = A(16x32 bf16) * B(32x16 bf16) + C(16x16 f32)
  return __builtin_amdgcn_wmma_f32_16x16x32_bf16(
      /*neg_a=*/false, a, /*neg_b=*/false, b,
      /*c_mod=*/(short)0, c, /*reuse_a=*/false, /*reuse_b=*/false);
}

__device__ __forceinline__ v16bf cat8(v8bf lo, v8bf hi) {
  return __builtin_shufflevector(lo, hi, 0,1,2,3,4,5,6,7,8,9,10,11,12,13,14,15);
}

// A-matrix 16x32 bf16 fragment: lane holds row m = lane%16,
// element j -> K = 16*hl + j  (one contiguous 32-byte load).
__device__ __forceinline__ v16bf load_a_frag(const __bf16* rowPtr, int hl) {
  return *(const v16bf*)(rowPtr + hl * 16);
}

// B-matrix 32x16 bf16 fragment: lane holds col n = lane%16,
// element j -> K = 16*hl + j  (one contiguous 32-byte load).
__device__ __forceinline__ v16bf load_b_frag(const __bf16* colPtr, int hl) {
  return *(const v16bf*)(colPtr + hl * 16);
}

// ---- Pack kernels ---------------------------------------------------------

__global__ void pack_x_bf16(const float* __restrict__ x,
                            __bf16* __restrict__ y, int n) {
  int i = blockIdx.x * blockDim.x + threadIdx.x;
  if (i < n) y[i] = (__bf16)x[i];
}

// wt[n][k] = w[k][n]  (transpose + bf16 convert)
__global__ void pack_w_t_bf16(const float* __restrict__ w,
                              __bf16* __restrict__ wt) {
  int i = blockIdx.x * blockDim.x + threadIdx.x;
  if (i < DM_ * DM_) {
    int n = i / DM_, k = i % DM_;
    wt[i] = (__bf16)w[k * DM_ + n];
  }
}

// ---- Fused QKV projection GEMM -------------------------------------------
// One wave computes a 32x64 output strip of one of {Q,K,V}: two 16-row A
// fragments amortize each B fragment (0.75 fragment-loads per WMMA).
// Q stored [b,h,s,d] bf16, PRE-SCALED by 1/sqrt(DH)*log2(e);
// K stored [b,h,s,d] bf16 ; V stored transposed [b,h,d,s] bf16.

__global__ __launch_bounds__(256)
void qkv_gemm(const __bf16* __restrict__ Xb,   // [4096][768]
              const __bf16* __restrict__ Wt,   // [3][768][768] (n-major)
              const float* __restrict__ bq, const float* __restrict__ bk,
              const float* __restrict__ bv,
              __bf16* __restrict__ Qo, __bf16* __restrict__ Ko,
              __bf16* __restrict__ Vt) {
  const int lane = threadIdx.x;
  const int wv   = blockIdx.x * 8 + threadIdx.y;      // 4608 waves
  const int mat  = wv / (128 * 12);                   // 0=Q 1=K 2=V
  const int rem  = wv % (128 * 12);
  const int t0   = (rem / 12) * 32;                   // token tile base
  const int ng   = rem % 12;                          // head index
  const int n0   = ng * 64;
  const int m    = lane & 15, hl = lane >> 4;

  const __bf16* W    = Wt + (size_t)mat * DM_ * DM_;
  const float*  bias = (mat == 0) ? bq : (mat == 1) ? bk : bv;
  const float   osc  = (mat == 0) ? SCALE_Q_ : 1.0f;  // fold softmax scale

  v8f acc[2][4] = {};
  for (int k0 = 0; k0 < DM_; k0 += 32) {
    v16bf a0 = load_a_frag(Xb + (size_t)(t0 + m) * DM_ + k0, hl);
    v16bf a1 = load_a_frag(Xb + (size_t)(t0 + 16 + m) * DM_ + k0, hl);
#pragma unroll
    for (int c = 0; c < 4; ++c) {
      v16bf b = load_b_frag(W + (size_t)(n0 + c * 16 + m) * DM_ + k0, hl);
      acc[0][c] = wmma_bf16(a0, b, acc[0][c]);
      acc[1][c] = wmma_bf16(a1, b, acc[1][c]);
    }
  }

#pragma unroll
  for (int u = 0; u < 2; ++u) {
#pragma unroll
    for (int c = 0; c < 4; ++c) {
      const int d  = c * 16 + m;                 // dim within head
      const float bvl = bias[n0 + d];
#pragma unroll
      for (int r = 0; r < 8; ++r) {
        const int t  = t0 + u * 16 + r + 8 * hl; // token
        const int bb = t / S_, s = t % S_;
        const int bh = bb * H_ + ng;
        const float val = (acc[u][c][r] + bvl) * osc;
        if (mat == 2)
          Vt[((size_t)bh * DH_ + d) * S_ + s] = (__bf16)val;    // transposed
        else if (mat == 0)
          Qo[((size_t)bh * S_ + s) * DH_ + d] = (__bf16)val;
        else
          Ko[((size_t)bh * S_ + s) * DH_ + d] = (__bf16)val;
      }
    }
  }
}

// ---- Flash attention ------------------------------------------------------
// One wave owns 32 queries (two 16-row subtiles) of one (b,h); streams keys
// 32 at a time with online softmax in the exp2 domain. K/V fragments are
// shared across the two query subtiles. P is transposed D->A layout via a
// private 2KB LDS slice; row-sums (softmax denominator) are accumulated by
// an extra WMMA against an all-ones B fragment — no sum shuffles.

__global__ __launch_bounds__(256)
void attention(const __bf16* __restrict__ Qb, const __bf16* __restrict__ Kb,
               const __bf16* __restrict__ Vt, const float* __restrict__ mask,
               float* __restrict__ out) {
  __shared__ __align__(32) __bf16 Plds[8][32][32];     // 16 KB

  const int lane = threadIdx.x, wid = threadIdx.y;
  const int wv    = blockIdx.x * 8 + wid;              // 1536 waves
  const int qTile = wv % (S_ / 32);
  const int bh    = wv / (S_ / 32);
  const int h     = bh % H_, b = bh / H_;
  const int q0    = qTile * 32;
  const int m     = lane & 15, hl = lane >> 4;

  const __bf16* Qh = Qb + (size_t)bh * S_ * DH_;
  const __bf16* Kh = Kb + (size_t)bh * S_ * DH_;
  const __bf16* Vh = Vt + (size_t)bh * DH_ * S_;
  const float*  mk = mask + (size_t)b * S_;

  v16bf qa[2][2];
#pragma unroll
  for (int u = 0; u < 2; ++u) {
    qa[u][0] = load_a_frag(Qh + (size_t)(q0 + u * 16 + m) * DH_ + 0,  hl);
    qa[u][1] = load_a_frag(Qh + (size_t)(q0 + u * 16 + m) * DH_ + 32, hl);
  }

  // all-ones B fragment for row-sum WMMAs
  v16bf onesb;
#pragma unroll
  for (int j = 0; j < 16; ++j) onesb[j] = (__bf16)1.0f;

  float mrow[2][8];                       // running row max (exp2 domain)
#pragma unroll
  for (int u = 0; u < 2; ++u)
#pragma unroll
    for (int r = 0; r < 8; ++r) mrow[u][r] = -__builtin_inff();
  v8f acc[2][4] = {};                     // ctx accumulators
  v8f lacc[2]   = {};                     // row-sum accumulators (all cols eq)

  for (int k0 = 0; k0 < S_; k0 += 32) {
    if (k0 + 32 < S_) {                   // global_prefetch_b8 of next chunk
      __builtin_prefetch(Kh + (size_t)(k0 + 32 + m) * DH_, 0, 0);
      __builtin_prefetch(Vh + (size_t)m * S_ + k0 + 32, 0, 0);
    }
    // ---- K fragments for this 32-key step (shared by both query subtiles)
    v16bf kb00 = load_b_frag(Kh + (size_t)(k0 + m) * DH_ + 0,       hl);
    v16bf kb01 = load_b_frag(Kh + (size_t)(k0 + m) * DH_ + 32,      hl);
    v16bf kb10 = load_b_frag(Kh + (size_t)(k0 + 16 + m) * DH_ + 0,  hl);
    v16bf kb11 = load_b_frag(Kh + (size_t)(k0 + 16 + m) * DH_ + 32, hl);
    const float mkl0 = mk[k0 + m]      * LOG2E_;   // mask -> exp2 domain
    const float mkl1 = mk[k0 + 16 + m] * LOG2E_;

    float alpha[2][8];
#pragma unroll
    for (int u = 0; u < 2; ++u) {
      // ---- scores (already in exp2 domain: Q pre-scaled) 16q x 32k
      v8f s0 = {}, s1 = {};
      s0 = wmma_bf16(qa[u][0], kb00, s0);
      s0 = wmma_bf16(qa[u][1], kb01, s0);
      s1 = wmma_bf16(qa[u][0], kb10, s1);
      s1 = wmma_bf16(qa[u][1], kb11, s1);

      // ---- online max + exp2 (rows live across 16-lane halves)
#pragma unroll
      for (int r = 0; r < 8; ++r) {
        const float a0 = s0[r] + mkl0;
        const float a1 = s1[r] + mkl1;
        float rm = fmaxf(a0, a1);
        rm = fmaxf(rm, __shfl_xor(rm, 1, 32));
        rm = fmaxf(rm, __shfl_xor(rm, 2, 32));
        rm = fmaxf(rm, __shfl_xor(rm, 4, 32));
        rm = fmaxf(rm, __shfl_xor(rm, 8, 32));
        const float mn = fmaxf(mrow[u][r], rm);
        alpha[u][r] = exp2f(mrow[u][r] - mn);
        mrow[u][r]  = mn;
        const float p0 = exp2f(a0 - mn);
        const float p1 = exp2f(a1 - mn);
        // ---- D-layout -> A-layout transpose of P via per-wave LDS slice
        const int row = u * 16 + r + 8 * hl;
        Plds[wid][row][m]      = (__bf16)p0;
        Plds[wid][row][16 + m] = (__bf16)p1;
      }
      // ---- rescale ctx + row-sum accumulators by alpha
      lacc[u] *= *(const v8f*)alpha[u];
#pragma unroll
      for (int c = 0; c < 4; ++c)
#pragma unroll
        for (int r = 0; r < 8; ++r) acc[u][c][r] *= alpha[u][r];
    }
    asm volatile("s_wait_dscnt 0" ::: "memory");
    const v16bf pa0 = load_a_frag(&Plds[wid][m][0],      hl);
    const v16bf pa1 = load_a_frag(&Plds[wid][16 + m][0], hl);

    // ---- row sums via WMMA: lacc += P x ones (no shuffle reductions)
    lacc[0] = wmma_bf16(pa0, onesb, lacc[0]);
    lacc[1] = wmma_bf16(pa1, onesb, lacc[1]);

    // ---- ctx += P(32x32) x V(32x16) over the 4 dim-chunks of DH
#pragma unroll
    for (int c = 0; c < 4; ++c) {
      v16bf vb = load_b_frag(Vh + (size_t)(c * 16 + m) * S_ + k0, hl);
      acc[0][c] = wmma_bf16(pa0, vb, acc[0][c]);
      acc[1][c] = wmma_bf16(pa1, vb, acc[1][c]);
    }
  }

  // ---- normalize + scatter to [B,S,DM] f32
#pragma unroll
  for (int u = 0; u < 2; ++u) {
    float inv[8];
#pragma unroll
    for (int r = 0; r < 8; ++r) inv[r] = 1.0f / lacc[u][r];
#pragma unroll
    for (int c = 0; c < 4; ++c) {
      const int d = c * 16 + m;
#pragma unroll
      for (int r = 0; r < 8; ++r) {
        const int q = q0 + u * 16 + r + 8 * hl;
        out[((size_t)(b * S_ + q)) * DM_ + h * DH_ + d] = acc[u][c][r] * inv[r];
      }
    }
  }
}

// ---- launcher -------------------------------------------------------------

extern "C" void kernel_launch(void* const* d_in, const int* in_sizes, int n_in,
                              void* d_out, int out_size, void* d_ws,
                              size_t ws_size, hipStream_t stream) {
  const float* X    = (const float*)d_in[0];
  const float* mask = (const float*)d_in[1];
  const float* Wq   = (const float*)d_in[2];
  const float* bq   = (const float*)d_in[3];
  const float* Wk   = (const float*)d_in[4];
  const float* bk   = (const float*)d_in[5];
  const float* Wv   = (const float*)d_in[6];
  const float* bv   = (const float*)d_in[7];
  float* out = (float*)d_out;

  char* ws = (char*)d_ws;
  __bf16* Xb = (__bf16*)ws; ws += (size_t)NT_ * DM_ * 2;       // 6 MB
  __bf16* Wt = (__bf16*)ws; ws += (size_t)3 * DM_ * DM_ * 2;   // 3.5 MB
  __bf16* Qb = (__bf16*)ws; ws += (size_t)NT_ * DM_ * 2;       // 6 MB
  __bf16* Kb = (__bf16*)ws; ws += (size_t)NT_ * DM_ * 2;       // 6 MB
  __bf16* Vt = (__bf16*)ws; ws += (size_t)NT_ * DM_ * 2;       // 6 MB

  {
    int n = NT_ * DM_;
    pack_x_bf16<<<(n + 255) / 256, 256, 0, stream>>>(X, Xb, n);
  }
  const int nw = (DM_ * DM_ + 255) / 256;
  pack_w_t_bf16<<<nw, 256, 0, stream>>>(Wq, Wt + 0 * DM_ * DM_);
  pack_w_t_bf16<<<nw, 256, 0, stream>>>(Wk, Wt + 1 * DM_ * DM_);
  pack_w_t_bf16<<<nw, 256, 0, stream>>>(Wv, Wt + 2 * DM_ * DM_);

  dim3 blk(32, 8);                                   // 8 waves (wave32)
  qkv_gemm<<<(3 * 128 * 12) / 8, blk, 0, stream>>>(Xb, Wt, bq, bk, bv,
                                                   Qb, Kb, Vt);
  attention<<<(B_ * H_ * (S_ / 32)) / 8, blk, 0, stream>>>(Qb, Kb, Vt, mask,
                                                           out);
}
